// FeedForward_7335804141846
// MI455X (gfx1250) — 8-bit matrix operand order repaired and checked by reading, compiled, not run on hardware
//
#include <hip/hip_runtime.h>
#include <hip/hip_bf16.h>
#include <math.h>


typedef __attribute__((ext_vector_type(8)))  int      v8i;
typedef __attribute__((ext_vector_type(4)))  int      v4i;
typedef __attribute__((ext_vector_type(4)))  unsigned v4u;

#define EPSQ 1e-5f

#if __has_builtin(__builtin_amdgcn_tensor_load_to_lds)
#define HAVE_TDM 1
#endif

__device__ __forceinline__ float blk_sum(float v, float* red) {
    int t = threadIdx.x;
    red[t] = v; __syncthreads();
    for (int off = 128; off > 0; off >>= 1) {
        if (t < off) red[t] += red[t + off];
        __syncthreads();
    }
    float r = red[0]; __syncthreads();
    return r;
}

__device__ __forceinline__ float blk_max(float v, float* red) {
    int t = threadIdx.x;
    red[t] = v; __syncthreads();
    for (int off = 128; off > 0; off >>= 1) {
        if (t < off) red[t] = fmaxf(red[t], red[t + off]);
        __syncthreads();
    }
    float r = red[0]; __syncthreads();
    return r;
}

__global__ __launch_bounds__(256)
void absum_kernel(const float* __restrict__ w, long long n, float* __restrict__ out_sum) {
    __shared__ float red[256];
    float s = 0.0f;
    for (long long i = (long long)blockIdx.x * blockDim.x + threadIdx.x; i < n;
         i += (long long)gridDim.x * blockDim.x)
        s += fabsf(w[i]);
    float tot = blk_sum(s, red);
    if (threadIdx.x == 0) atomicAdd(out_sum, tot);
}

__global__ void wscale_finalize_kernel(float* f, long long n1, long long n2) {
    if (threadIdx.x == 0 && blockIdx.x == 0) {
        f[1] = 1.0f / fmaxf(f[0] / (float)n1, EPSQ);
        f[3] = 1.0f / fmaxf(f[2] / (float)n2, EPSQ);
    }
}

__global__ __launch_bounds__(256)
void wquant_kernel(const float* __restrict__ w, signed char* __restrict__ wq,
                   long long n, const float* __restrict__ swp) {
    float sw = *swp;
    long long i = (long long)blockIdx.x * blockDim.x + threadIdx.x;
    if (i < n) {
        float v = rintf(w[i] * sw);
        v = fminf(fmaxf(v, -1.0f), 1.0f);
        wq[i] = (signed char)v;
    }
}

__global__ __launch_bounds__(256)
void ln_quant_kernel(const float* __restrict__ x, const float* __restrict__ gamma,
                     const float* __restrict__ beta, signed char* __restrict__ xq,
                     float* __restrict__ sx, int D) {
    __shared__ float red[256];
    long long tok = blockIdx.x;
    const float* row = x + tok * (long long)D;

    float s = 0.0f;
    for (int i = threadIdx.x; i < D; i += 256) s += row[i];
    float mu = blk_sum(s, red) / (float)D;

    float v = 0.0f;
    for (int i = threadIdx.x; i < D; i += 256) { float d = row[i] - mu; v += d * d; }
    float var = blk_sum(v, red) / (float)D;
    float rstd = rsqrtf(var + EPSQ);

    float amax = 0.0f;
    for (int i = threadIdx.x; i < D; i += 256) {
        float y = (row[i] - mu) * rstd * gamma[i] + beta[i];
        amax = fmaxf(amax, fabsf(y));
    }
    amax = blk_max(amax, red);
    float scale = 127.0f / fmaxf(amax, EPSQ);

    signed char* qrow = xq + tok * (long long)D;
    for (int i = threadIdx.x; i < D; i += 256) {
        float y = (row[i] - mu) * rstd * gamma[i] + beta[i];
        float q = fminf(fmaxf(rintf(y * scale), -128.0f), 127.0f);
        qrow[i] = (signed char)q;
    }
    if (threadIdx.x == 0) sx[tok] = scale;
}

__global__ __launch_bounds__(256)
void rowquant_kernel(const float* __restrict__ h, signed char* __restrict__ hq,
                     float* __restrict__ sh, int D) {
    __shared__ float red[256];
    long long tok = blockIdx.x;
    const float* row = h + tok * (long long)D;

    float amax = 0.0f;
    for (int i = threadIdx.x; i < D; i += 256) amax = fmaxf(amax, fabsf(row[i]));
    amax = blk_max(amax, red);
    float scale = 127.0f / fmaxf(amax, EPSQ);

    signed char* qrow = hq + tok * (long long)D;
    for (int i = threadIdx.x; i < D; i += 256) {
        float q = fminf(fmaxf(rintf(row[i] * scale), -128.0f), 127.0f);
        qrow[i] = (signed char)q;
    }
    if (threadIdx.x == 0) sh[tok] = scale;
}

#ifdef HAVE_TDM
__device__ __forceinline__ void tdm_load_tile_2d(const void* gsrc, unsigned lds_off,
                                                 unsigned tile_x, unsigned tile_y,
                                                 unsigned tensor_x, unsigned tensor_y,
                                                 unsigned long long stride_x) {
    unsigned long long ga = (unsigned long long)gsrc;
    v4u g0;
    g0[0] = 1u;
    g0[1] = lds_off;
    g0[2] = (unsigned)(ga & 0xFFFFFFFFu);
    g0[3] = (unsigned)((ga >> 32) & 0x1FFFFFFu) | (2u << 30);
    v8i g1;
    g1[0] = 0;
    g1[1] = (int)((tensor_x & 0xFFFFu) << 16);
    g1[2] = (int)(((tensor_x >> 16) & 0xFFFFu) | ((tensor_y & 0xFFFFu) << 16));
    g1[3] = (int)(((tensor_y >> 16) & 0xFFFFu) | ((tile_x & 0xFFFFu) << 16));
    g1[4] = (int)(tile_y & 0xFFFFu);
    g1[5] = (int)(unsigned)(stride_x & 0xFFFFFFFFull);
    g1[6] = (int)(unsigned)((stride_x >> 32) & 0xFFFFull);
    g1[7] = 0;
    v4i z4 = {0, 0, 0, 0};
#if defined(__clang_major__) && __clang_major__ >= 23
    v8i z8 = {0, 0, 0, 0, 0, 0, 0, 0};
    __builtin_amdgcn_tensor_load_to_lds(g0, g1, z4, z4, z8, 0);
#else
    __builtin_amdgcn_tensor_load_to_lds(g0, g1, z4, z4, 0);
#endif
}
#endif

__global__ __launch_bounds__(256)
void gemm_iu8_wmma_kernel(const signed char* __restrict__ A,
                          const signed char* __restrict__ B,
                          const float* __restrict__ bias,
                          const float* __restrict__ sx,
                          const float* __restrict__ swp,
                          float* __restrict__ C,
                          int M, int N, int K, int apply_gelu) {
    __shared__ __align__(16) signed char lA[2 * 64 * 64];
    __shared__ __align__(16) signed char lB[2 * 128 * 64];

    const float sw = *swp;
    const int tile_n0 = blockIdx.x * 128;
    const int tile_m0 = blockIdx.y * 64;
    const int tid  = threadIdx.x;
    const int lane = tid & 31;
    const int wave = tid >> 5;
    const int half = lane >> 4;
    const int lr   = lane & 15;
    const int wm   = wave & 3;
    const int wn   = wave >> 2;

    v8i acc[4];
#pragma unroll
    for (int nt = 0; nt < 4; ++nt) acc[nt] = (v8i){0, 0, 0, 0, 0, 0, 0, 0};

#ifdef HAVE_TDM
    if (wave == 0) {
        tdm_load_tile_2d(A + (long long)tile_m0 * K, (unsigned)(size_t)lA,
                         64u, 64u, (unsigned)K, (unsigned)M, (unsigned long long)K);
        tdm_load_tile_2d(B + (long long)tile_n0 * K, (unsigned)(size_t)lB,
                         64u, 128u, (unsigned)K, (unsigned)N, (unsigned long long)K);
    }
    int buf = 0;
    for (int k0 = 0; k0 < K; k0 += 64) {
        const int nb = buf ^ 1;
        const bool has_next = (k0 + 64) < K;
        if (wave == 0) {
            if (has_next) {
                tdm_load_tile_2d(A + (long long)tile_m0 * K + (k0 + 64),
                                 (unsigned)(size_t)(lA + nb * (64 * 64)),
                                 64u, 64u, (unsigned)K, (unsigned)M,
                                 (unsigned long long)K);
                tdm_load_tile_2d(B + (long long)tile_n0 * K + (k0 + 64),
                                 (unsigned)(size_t)(lB + nb * (128 * 64)),
                                 64u, 128u, (unsigned)K, (unsigned)N,
                                 (unsigned long long)K);
                __builtin_amdgcn_s_wait_tensorcnt(2);
            } else {
                __builtin_amdgcn_s_wait_tensorcnt(0);
            }
        }
        __syncthreads();
#else
    int buf = 0;
    for (int k0 = 0; k0 < K; k0 += 64) {
        {
            int arow = tid >> 2, acol = (tid & 3) << 4;
            *(int4*)(lA + arow * 64 + acol) =
                *(const int4*)(A + (long long)(tile_m0 + arow) * K + k0 + acol);
            int brow = tid >> 1, bcol = (tid & 1) << 5;
            *(int4*)(lB + brow * 64 + bcol) =
                *(const int4*)(B + (long long)(tile_n0 + brow) * K + k0 + bcol);
            *(int4*)(lB + brow * 64 + bcol + 16) =
                *(const int4*)(B + (long long)(tile_n0 + brow) * K + k0 + bcol + 16);
        }
        __syncthreads();
#endif
        const signed char* lAb = lA + buf * (64 * 64);
        const signed char* lBb = lB + buf * (128 * 64);

        v8i a;
        {
            const signed char* ar = lAb + (wm * 16 + lr) * 64;
#pragma unroll
            for (int p = 0; p < 4; ++p) {
                int2 v = *(const int2*)(ar + (p >> 1) * 32 + (half << 4) + (p & 1) * 8);
                a[2 * p]     = v.x;
                a[2 * p + 1] = v.y;
            }
        }
#pragma unroll
        for (int nt = 0; nt < 4; ++nt) {
            const signed char* br = lBb + (wn * 64 + nt * 16 + lr) * 64;
            int4 q0 = *(const int4*)(br + (half << 4));
            int4 q1 = *(const int4*)(br + 32 + (half << 4));
            v8i b;
            b[0] = q0.x; b[1] = q0.y; b[2] = q0.z; b[3] = q0.w;
            b[4] = q1.x; b[5] = q1.y; b[6] = q1.z; b[7] = q1.w;
            acc[nt] = __builtin_amdgcn_wmma_i32_16x16x64_iu8(true, a, true, b,
                                                             acc[nt], false, false);
        }
        __syncthreads();
        buf ^= 1;
    }

    float inv[8];
#pragma unroll
    for (int r = 0; r < 8; ++r)
        inv[r] = 1.0f / (sx[tile_m0 + wm * 16 + half * 8 + r] * sw);
#pragma unroll
    for (int nt = 0; nt < 4; ++nt) {
        int gn = tile_n0 + wn * 64 + nt * 16 + lr;
        float bi = bias[gn];
#pragma unroll
        for (int r = 0; r < 8; ++r) {
            int m = tile_m0 + wm * 16 + half * 8 + r;
            float y = (float)acc[nt][r] * inv[r] + bi;
            if (apply_gelu)
                y = 0.5f * y * (1.0f + erff(y * 0.70710678118654752440f));
            C[(long long)m * N + gn] = y;
        }
    }
}

extern "C" void kernel_launch(void* const* d_in, const int* in_sizes, int n_in,
                              void* d_out, int out_size, void* d_ws, size_t ws_size,
                              hipStream_t stream) {
    const float* x     = (const float*)d_in[0];
    const float* w1    = (const float*)d_in[1];
    const float* b1    = (const float*)d_in[2];
    const float* w2    = (const float*)d_in[3];
    const float* b2    = (const float*)d_in[4];
    const float* gamma = (const float*)d_in[5];
    const float* beta  = (const float*)d_in[6];

    const int dim    = in_sizes[4];
    const int hidden = in_sizes[2];
    const int Mtok   = in_sizes[0] / dim;
    const long long n_w1 = (long long)hidden * dim;
    const long long n_w2 = (long long)dim * hidden;

    char* base = (char*)d_ws;
    size_t off = 0;
    auto take = [&](size_t bytes) -> char* {
        char* p = base + off;
        off = (off + bytes + 255) & ~(size_t)255;
        return p;
    };
    float*       f   = (float*)take(16);
    float*       sx  = (float*)take((size_t)Mtok * 4);
    float*       sh  = (float*)take((size_t)Mtok * 4);
    signed char* xq  = (signed char*)take((size_t)Mtok * dim);
    signed char* wq1 = (signed char*)take((size_t)n_w1);
    signed char* wq2 = (signed char*)take((size_t)n_w2);
    signed char* hq  = (signed char*)take((size_t)Mtok * hidden);
    float*       h   = (float*)take((size_t)Mtok * hidden * 4);
    (void)ws_size;

    hipMemsetAsync(f, 0, 16, stream);
    absum_kernel<<<512, 256, 0, stream>>>(w1, n_w1, f + 0);
    absum_kernel<<<512, 256, 0, stream>>>(w2, n_w2, f + 2);
    wscale_finalize_kernel<<<1, 32, 0, stream>>>(f, n_w1, n_w2);

    wquant_kernel<<<(unsigned)((n_w1 + 255) / 256), 256, 0, stream>>>(w1, wq1, n_w1, f + 1);
    wquant_kernel<<<(unsigned)((n_w2 + 255) / 256), 256, 0, stream>>>(w2, wq2, n_w2, f + 3);

    ln_quant_kernel<<<Mtok, 256, 0, stream>>>(x, gamma, beta, xq, sx, dim);

    gemm_iu8_wmma_kernel<<<dim3(hidden / 128, Mtok / 64), 256, 0, stream>>>(
        xq, wq1, b1, sx, f + 1, h, Mtok, hidden, dim,  1);

    rowquant_kernel<<<Mtok, 256, 0, stream>>>(h, hq, sh, hidden);

    gemm_iu8_wmma_kernel<<<dim3(dim / 128, Mtok / 64), 256, 0, stream>>>(
        hq, wq2, b2, sh, f + 3, (float*)d_out, Mtok, dim, hidden,  0);
}
